// MultiHeadAttention_62105227100794
// MI455X (gfx1250) — compile-verified, matrix operand order checked by reading, not run on hardware
//
#include <hip/hip_runtime.h>

// MHA: B=4, S=4096, D=256, H=4, depth=64. fp32 in/out, bf16 WMMA compute.
// d_in order: 0=v, 1=k, 2=q, 3=mask, 4=wq_w, 5=wq_b, 6=wk_w, 7=wk_b,
//             8=wv_w, 9=wv_b, 10=wo_w, 11=wo_b
// Workspace: Qh(8MB) Kh(8MB) Vt(8MB, transposed) Att(8MB) bf16.

#define BDIM 4
#define SDIM 4096
#define DDIM 256
#define HDIM 4
#define DEPTH 64

typedef __attribute__((ext_vector_type(16))) __bf16 v16bf;
typedef __attribute__((ext_vector_type(8)))  float  v8f;

static __device__ __forceinline__ v8f v8f_zero() {
  v8f z;
#pragma unroll
  for (int i = 0; i < 8; ++i) z[i] = 0.0f;
  return z;
}

// 16x32 bf16 fragment from a row-major (leading-dim = stride) source.
// ISA A layout: lanes 0-15 -> row=lane, K {0..7,16..23}; lanes 16-31 -> row=lane-16,
// K {8..15,24..31}. Identical per-lane pattern serves as the B-matrix loader when
// the source is N-major (holds B^T), which is how all tiles are staged.
static __device__ __forceinline__ v16bf load_frag_rowmajor(const __bf16* p, int stride) {
  const int lane = threadIdx.x & 31;
  const int row  = lane & 15;
  const int half = lane >> 4;
  const __bf16* r = p + row * stride + half * 8;
  v16bf a;
#pragma unroll
  for (int i = 0; i < 8; ++i) a[i] = r[i];
#pragma unroll
  for (int i = 0; i < 8; ++i) a[8 + i] = r[16 + i];
  return a;
}

static __device__ __forceinline__ v8f wmma_bf16(v16bf a, v16bf b, v8f c) {
  return __builtin_amdgcn_wmma_f32_16x16x32_bf16(false, a, false, b, (short)0, c,
                                                 false, false);
}

// Async 16-byte global->LDS copy (per lane), tracked by ASYNCcnt.
// Low 32 bits of a generic LDS pointer are the wave-relative LDS byte address.
static __device__ __forceinline__ void async_b128(const void* g, const void* l) {
  const unsigned lds = (unsigned)(unsigned long long)l;
  asm volatile("global_load_async_to_lds_b128 %0, %1, off"
               :: "v"(lds), "v"(g)
               : "memory");
}
static __device__ __forceinline__ void async_wait_all() {
  asm volatile("s_wait_asynccnt 0" ::: "memory");
}

// ---------------------------------------------------------------------------
// GEMM + bias: Y[N=16384, 256] = X @ W + b
// IN_BF16: X is bf16 (attention output) else fp32.
// OUT_MODE: 0 = fp32 row-major [N,256]
//           1 = bf16 head-split [B,H,S,64]
//           2 = bf16 head-split transposed [B,H,64,S]   (for V)
// Block: 256 threads (8 waves). Tile: 128 rows x 64 cols; K staged 32 at a time.
// Grid: 512 blocks; rb = blockIdx.x>>2, cb = blockIdx.x&3.
// ---------------------------------------------------------------------------
template <bool IN_BF16, int OUT_MODE>
__global__ __launch_bounds__(256) void gemm_bias_kernel(
    const void* __restrict__ xin, const float* __restrict__ w,
    const float* __restrict__ bias, void* __restrict__ outp) {
  __shared__ __bf16 Xl[128][32];   // X tile, row-major
  __shared__ __bf16 Wl[64][32];    // W tile, N-major (transposed)

  const int tid  = threadIdx.x;
  const int wave = tid >> 5;
  const int lane = tid & 31;
  const int half = lane >> 4;
  const int col  = lane & 15;
  const int rb   = blockIdx.x >> 2;
  const int cb   = blockIdx.x & 3;
  const int row0 = rb * 128;
  const int col0 = cb * 64;

  v8f acc[4];
#pragma unroll
  for (int nt = 0; nt < 4; ++nt) acc[nt] = v8f_zero();

  for (int k0 = 0; k0 < DDIM; k0 += 32) {
    __syncthreads();
    // Stage X tile [128 x 32].
    {
      const int r = tid >> 1;
      const int c = (tid & 1) * 16;
      if (!IN_BF16) {
        const float4* src = (const float4*)((const float*)xin +
                                            (size_t)(row0 + r) * DDIM + k0 + c);
#pragma unroll
        for (int q = 0; q < 4; ++q) {
          const float4 f = src[q];
          Xl[r][c + 4 * q + 0] = (__bf16)f.x;
          Xl[r][c + 4 * q + 1] = (__bf16)f.y;
          Xl[r][c + 4 * q + 2] = (__bf16)f.z;
          Xl[r][c + 4 * q + 3] = (__bf16)f.w;
        }
      } else {
        const uint4* src = (const uint4*)((const __bf16*)xin +
                                          (size_t)(row0 + r) * DDIM + k0 + c);
        *(uint4*)&Xl[r][c]     = src[0];
        *(uint4*)&Xl[r][c + 8] = src[1];
      }
    }
    // Stage W tile transposed: Wl[n][k].  Coalesced global reads,
    // contiguous 16B LDS store per thread.
    {
      const int c  = tid & 63;        // column (n)
      const int rg = tid >> 6;        // 0..3 -> rows rg*8 .. rg*8+7
      const float* src = w + (size_t)(k0 + rg * 8) * DDIM + col0 + c;
      __bf16 tmp[8];
#pragma unroll
      for (int i = 0; i < 8; ++i) tmp[i] = (__bf16)src[(size_t)i * DDIM];
#pragma unroll
      for (int i = 0; i < 8; ++i) Wl[c][rg * 8 + i] = tmp[i];
    }
    __syncthreads();

    v16bf a = load_frag_rowmajor(&Xl[wave * 16][0], 32);
#pragma unroll
    for (int nt = 0; nt < 4; ++nt) {
      v16bf bb = load_frag_rowmajor(&Wl[nt * 16][0], 32);
      acc[nt] = wmma_bf16(a, bb, acc[nt]);
    }
  }

  // Epilogue. C layout: element j -> row m = j + 8*half, col = lane&15.
  const int nrow0 = row0 + wave * 16;
#pragma unroll
  for (int nt = 0; nt < 4; ++nt) {
    const int c = col0 + nt * 16 + col;
    const float bv = bias[c];
#pragma unroll
    for (int j = 0; j < 8; ++j) {
      const int m = j + 8 * half;
      const int n = nrow0 + m;
      const float val = acc[nt][j] + bv;
      const int b = n >> 12;            // / S
      const int s = n & (SDIM - 1);
      const int h = c >> 6;
      const int d = c & 63;
      if (OUT_MODE == 0) {
        ((float*)outp)[(size_t)n * DDIM + c] = val;
      } else if (OUT_MODE == 1) {
        ((__bf16*)outp)[((((size_t)b * HDIM + h) * SDIM + s) << 6) + d] = (__bf16)val;
      } else {
        ((__bf16*)outp)[(((size_t)b * HDIM + h) * DEPTH + d) * SDIM + s] = (__bf16)val;
      }
    }
  }
}

// ---------------------------------------------------------------------------
// Flash attention: per (b,h) and 128-query tile. 8 waves x 16 query rows each.
// Key tiles of 64; QK^T and P@V via bf16 WMMA; online softmax in f32.
// K/V tiles double-buffered with async global->LDS copies (ASYNCcnt):
//   iter kt: issue async -> buf^1, compute on buf, s_wait_asynccnt 0, barrier.
// Grid: 512 blocks; bh = blockIdx.x>>5, qt = blockIdx.x&31.
// ---------------------------------------------------------------------------
__global__ __launch_bounds__(256) void attn_kernel(
    const void* __restrict__ Qp, const void* __restrict__ Kp,
    const void* __restrict__ Vp, const float* __restrict__ mask,
    void* __restrict__ attp) {
  __shared__ __bf16 Ql[128][64];        // 16 KB, resident (A source)
  __shared__ __bf16 Kl[2][64][64];      // 16 KB, double-buffered (N-major for QK^T)
  __shared__ __bf16 Vl[2][64][64];      // 16 KB, double-buffered (N-major for P@V)
  __shared__ __bf16 Pl[8][16][64];      // 16 KB, per-wave P strips

  const __bf16* Qh = (const __bf16*)Qp;
  const __bf16* Kh = (const __bf16*)Kp;
  const __bf16* Vt = (const __bf16*)Vp;
  __bf16* att = (__bf16*)attp;

  const int tid  = threadIdx.x;
  const int wave = tid >> 5;
  const int lane = tid & 31;
  const int half = lane >> 4;
  const int col  = lane & 15;
  const int qt   = blockIdx.x & 31;
  const int bh   = blockIdx.x >> 5;           // b*H + h
  const int b    = bh >> 2;                   // H = 4
  const int h    = bh & 3;
  const size_t bh_base = (size_t)bh * SDIM * DEPTH;   // same extent for Kh and Vt
  const float* mrow = mask + (size_t)b * SDIM;

  const int r64 = tid >> 2;            // 0..63 (staging row)
  const int c64 = (tid & 3) * 16;      // 0,16,32,48

  // Per-thread async staging of one K tile + one V^T tile into buffer `buf`.
  auto stage_kv = [&](int kt, int buf) {
    const __bf16* ks = Kh + bh_base + (size_t)(kt * 64 + r64) * DEPTH + c64;
    const __bf16* vs = Vt + bh_base + (size_t)r64 * SDIM + kt * 64 + c64;
    async_b128(ks,     &Kl[buf][r64][c64]);
    async_b128(ks + 8, &Kl[buf][r64][c64 + 8]);
    async_b128(vs,     &Vl[buf][r64][c64]);
    async_b128(vs + 8, &Vl[buf][r64][c64 + 8]);
  };

  // Prologue: stage Q tile [128 x 64] and K/V tile 0 together.
  {
    const int r = tid >> 1;
    const int c = (tid & 1) * 32;
    const __bf16* src = Qh + bh_base + (size_t)(qt * 128 + r) * DEPTH + c;
#pragma unroll
    for (int q = 0; q < 4; ++q) async_b128(src + 8 * q, &Ql[r][c + 8 * q]);
  }
  stage_kv(0, 0);
  async_wait_all();
  __syncthreads();

  v16bf aQ[2];
  aQ[0] = load_frag_rowmajor(&Ql[wave * 16][0], 64);
  aQ[1] = load_frag_rowmajor(&Ql[wave * 16][32], 64);

  v8f acc[4];
#pragma unroll
  for (int nt = 0; nt < 4; ++nt) acc[nt] = v8f_zero();
  float mrun[8], lrun[8];
#pragma unroll
  for (int j = 0; j < 8; ++j) { mrun[j] = -3.0e38f; lrun[j] = 0.0f; }

  const int NT = SDIM / 64;
  for (int kt = 0; kt < NT; ++kt) {
    const int cur = kt & 1;
    // Overlap: fetch next tile into the other buffer while computing this one.
    // Safe: iteration kt-1 ended with wait+barrier after its reads of buf cur^1.
    if (kt + 1 < NT) {
      stage_kv(kt + 1, cur ^ 1);
      if (kt + 2 < NT) {
        // Warm L2 one more tile ahead.
        __builtin_prefetch(Kh + bh_base + (size_t)((kt + 2) * 64 + r64) * DEPTH + c64, 0, 0);
        __builtin_prefetch(Vt + bh_base + (size_t)r64 * SDIM + (kt + 2) * 64 + c64, 0, 0);
      }
    }

    // Scores: S = (Q K^T) * 0.125 + mask * -1e9.  4 col-tiles x 2 K-chunks.
    v8f sc[4];
#pragma unroll
    for (int nt = 0; nt < 4; ++nt) {
      v8f s = v8f_zero();
#pragma unroll
      for (int kc = 0; kc < 2; ++kc) {
        v16bf bb = load_frag_rowmajor(&Kl[cur][nt * 16][kc * 32], 64);
        s = wmma_bf16(aQ[kc], bb, s);
      }
      const float mv = mrow[kt * 64 + nt * 16 + col] * (-1.0e9f);
#pragma unroll
      for (int j = 0; j < 8; ++j) s[j] = s[j] * 0.125f + mv;
      sc[nt] = s;
    }

    // Online softmax per row (row m = j + 8*half; cols across 16 lanes).
#pragma unroll
    for (int j = 0; j < 8; ++j) {
      float v = fmaxf(fmaxf(sc[0][j], sc[1][j]), fmaxf(sc[2][j], sc[3][j]));
      v = fmaxf(v, __shfl_xor(v, 8, 32));
      v = fmaxf(v, __shfl_xor(v, 4, 32));
      v = fmaxf(v, __shfl_xor(v, 2, 32));
      v = fmaxf(v, __shfl_xor(v, 1, 32));
      const float mnew = fmaxf(mrun[j], v);
      const float corr = __expf(mrun[j] - mnew);
      mrun[j] = mnew;
      float rs = 0.0f;
#pragma unroll
      for (int nt = 0; nt < 4; ++nt) {
        const float p = __expf(sc[nt][j] - mnew);
        sc[nt][j] = p;
        rs += p;
      }
      rs += __shfl_xor(rs, 8, 32);
      rs += __shfl_xor(rs, 4, 32);
      rs += __shfl_xor(rs, 2, 32);
      rs += __shfl_xor(rs, 1, 32);
      lrun[j] = lrun[j] * corr + rs;
#pragma unroll
      for (int nt = 0; nt < 4; ++nt) acc[nt][j] *= corr;
    }

    // C-layout -> A-layout via per-wave LDS strip (in-order within wave).
#pragma unroll
    for (int nt = 0; nt < 4; ++nt)
#pragma unroll
      for (int j = 0; j < 8; ++j)
        Pl[wave][j + 8 * half][nt * 16 + col] = (__bf16)sc[nt][j];

    v16bf aP0 = load_frag_rowmajor(&Pl[wave][0][0], 64);
    v16bf aP1 = load_frag_rowmajor(&Pl[wave][0][32], 64);

    // acc += P @ V: B[k][n] = V[k][n] = Vl[n][k] (N-major source, contiguous).
#pragma unroll
    for (int nt = 0; nt < 4; ++nt) {
      v16bf b0 = load_frag_rowmajor(&Vl[cur][nt * 16][0], 64);
      acc[nt] = wmma_bf16(aP0, b0, acc[nt]);
      v16bf b1 = load_frag_rowmajor(&Vl[cur][nt * 16][32], 64);
      acc[nt] = wmma_bf16(aP1, b1, acc[nt]);
    }

    // Drain this iteration's async copies (next tile) and sync the workgroup.
    async_wait_all();
    __syncthreads();
  }

  // Epilogue: normalize, merge heads: att[(b*S+s)*256 + h*64 + d].
  const int qrow0 = qt * 128 + wave * 16;
#pragma unroll
  for (int j = 0; j < 8; ++j) {
    const float inv = 1.0f / lrun[j];
    const int m = j + 8 * half;
    const size_t rowoff = ((size_t)b * SDIM + (qrow0 + m)) * DDIM + h * DEPTH;
#pragma unroll
    for (int nt = 0; nt < 4; ++nt)
      att[rowoff + nt * 16 + col] = (__bf16)(acc[nt][j] * inv);
  }
}

extern "C" void kernel_launch(void* const* d_in, const int* in_sizes, int n_in,
                              void* d_out, int out_size, void* d_ws, size_t ws_size,
                              hipStream_t stream) {
  const float* v_in = (const float*)d_in[0];
  const float* k_in = (const float*)d_in[1];
  const float* q_in = (const float*)d_in[2];
  const float* mask = (const float*)d_in[3];
  const float* wq_w = (const float*)d_in[4];
  const float* wq_b = (const float*)d_in[5];
  const float* wk_w = (const float*)d_in[6];
  const float* wk_b = (const float*)d_in[7];
  const float* wv_w = (const float*)d_in[8];
  const float* wv_b = (const float*)d_in[9];
  const float* wo_w = (const float*)d_in[10];
  const float* wo_b = (const float*)d_in[11];

  char* ws = (char*)d_ws;
  void* Qh  = (void*)(ws);                         // 8 MB bf16 [B,H,S,64]
  void* Kh  = (void*)(ws + ((size_t)8 << 20));     // 8 MB bf16 [B,H,S,64]
  void* Vt  = (void*)(ws + ((size_t)16 << 20));    // 8 MB bf16 [B,H,64,S]
  void* Att = (void*)(ws + ((size_t)24 << 20));    // 8 MB bf16 [B,S,256]

  // Projections (fp32 in -> bf16 out; V transposed for depth-major staging).
  gemm_bias_kernel<false, 1><<<512, 256, 0, stream>>>(q_in, wq_w, wq_b, Qh);
  gemm_bias_kernel<false, 1><<<512, 256, 0, stream>>>(k_in, wk_w, wk_b, Kh);
  gemm_bias_kernel<false, 2><<<512, 256, 0, stream>>>(v_in, wv_w, wv_b, Vt);

  // Flash attention.
  attn_kernel<<<512, 256, 0, stream>>>(Qh, Kh, Vt, mask, Att);

  // Output projection (bf16 in -> fp32 out).
  gemm_bias_kernel<true, 0><<<512, 256, 0, stream>>>(Att, wo_w, wo_b, d_out);
}
